// TritonAdaptivePiecewiseLinear_88519275970723
// MI455X (gfx1250) — compile-verified, matrix operand order checked by reading, not run on hardware
//
#include <hip/hip_runtime.h>

typedef _Float16 v16h __attribute__((ext_vector_type(16)));
typedef _Float16 v8h  __attribute__((ext_vector_type(8)));
typedef float    v8f  __attribute__((ext_vector_type(8)));

#define B_ 128
#define I_ 512
#define O_ 512
#define P_ 16
#define K_ (I_ * P_) /* 8192 */

// ---------------------------------------------------------------------------
// Kernel 1: build the hat-basis matrix A (B_ x K_) in f16.
// One thread per (b, i). basis[b, i*16+p] = hat weight of breakpoint p at x[b,i].
// ---------------------------------------------------------------------------
__global__ void build_basis(const float* __restrict__ x,
                            const float* __restrict__ positions,
                            _Float16* __restrict__ A) {
  int t = blockIdx.x * blockDim.x + threadIdx.x;      // B_*I_ threads
  int b = t >> 9;                                     // / I_
  int i = t & (I_ - 1);

  float xv = x[(size_t)b * I_ + i];
  float xa = xv < 0.f ? xv : -xv;                     // anti-periodic fold: -|x|
  float xc = fminf(fmaxf(xa, -1.f), 1.f);

  // positions are shared across outputs: read positions[i, 0, :]
  float pos[P_];
  const float* psrc = positions + (size_t)i * O_ * P_;
#pragma unroll
  for (int p = 0; p < P_; ++p) pos[p] = psrc[p];

  // searchsorted(side='right'): count of pos[p] <= xc, then clip(idx-1, 0, P-2)
  int cnt = 0;
#pragma unroll
  for (int p = 0; p < P_; ++p) cnt += (pos[p] <= xc) ? 1 : 0;
  int idx = cnt - 1;
  idx = idx < 0 ? 0 : (idx > P_ - 2 ? P_ - 2 : idx);

  // branch-free select of lp/rp (keeps pos[] in registers, no scratch)
  float lp = pos[0], rp = pos[1];
#pragma unroll
  for (int p = 1; p <= P_ - 2; ++p) {
    bool sel = (p == idx);
    lp = sel ? pos[p] : lp;
    rp = sel ? pos[p + 1] : rp;
  }
  float denom = rp - lp;
  float tt = (denom != 0.f) ? (xc - lp) / denom : 0.f;
  tt = fminf(fmaxf(tt, 0.f), 1.f);

  v8h lo, hi;
#pragma unroll
  for (int p = 0; p < 8; ++p) {
    float w = (p == idx) ? (1.f - tt) : ((p == idx + 1) ? tt : 0.f);
    lo[p] = (_Float16)w;
  }
#pragma unroll
  for (int p = 8; p < 16; ++p) {
    float w = (p == idx) ? (1.f - tt) : ((p == idx + 1) ? tt : 0.f);
    hi[p - 8] = (_Float16)w;
  }
  v8h* dst = (v8h*)(A + (size_t)b * K_ + (size_t)i * P_);
  dst[0] = lo;
  dst[1] = hi;
}

// ---------------------------------------------------------------------------
// Kernel 2: transpose/convert values (I,O,P) f32 -> f16 B-matrix Bm[o][i*P+p]
// (column-major in N so WMMA B fragments are contiguous per lane).
// One thread per (i, o); o is the fast index -> coalesced 64B reads.
// ---------------------------------------------------------------------------
__global__ void pack_values(const float* __restrict__ values,
                            _Float16* __restrict__ Bm) {
  int t = blockIdx.x * blockDim.x + threadIdx.x;      // I_*O_ threads
  int i = t >> 9;
  int o = t & (O_ - 1);
  const float* src = values + ((size_t)i * O_ + o) * P_;
  v8h lo, hi;
#pragma unroll
  for (int p = 0; p < 8; ++p)  lo[p] = (_Float16)src[p];
#pragma unroll
  for (int p = 8; p < 16; ++p) hi[p - 8] = (_Float16)src[p];
  v8h* dst = (v8h*)(Bm + (size_t)o * K_ + (size_t)i * P_);
  dst[0] = lo;
  dst[1] = hi;
}

// ---------------------------------------------------------------------------
// Kernel 3: C(128x512) = A(128x8192) * Bm^T via v_wmma_f32_16x16x32_f16.
// One workgroup (4 waves) per 16x16 C tile; K split across the 4 waves
// (2048 each); 4 accumulator chains; explicit two-buffer software pipeline
// (no register rotation copies -> no forced loadcnt-0 drains);
// deterministic LDS reduction at the end.
// ---------------------------------------------------------------------------
__device__ __forceinline__ v16h load_frag(const _Float16* p) {
  v8h lo = *(const v8h*)p;
  v8h hi = *(const v8h*)(p + 16);
  return __builtin_shufflevector(lo, hi, 0, 1, 2, 3, 4, 5, 6, 7,
                                 8, 9, 10, 11, 12, 13, 14, 15);
}

#define WMMA_F16(a, b, c)                                                 \
  __builtin_amdgcn_wmma_f32_16x16x32_f16(false, (a), false, (b),          \
                                         (short)0, (c), false, false)

#define KSLICE (K_ / 4) /* 2048 per wave */

__global__ void wmma_gemm(const _Float16* __restrict__ A,
                          const _Float16* __restrict__ Bm,
                          float* __restrict__ C) {
  __shared__ v8f red[3][32];                 // partials from waves 1..3

  const int lane  = threadIdx.x & 31;
  const int wave  = threadIdx.x >> 5;        // 0..3
  const int tile  = blockIdx.x;              // 0..255
  const int mt    = tile & 7;                // 8 tiles along M
  const int nt    = tile >> 3;               // 32 tiles along N
  const int l15   = lane & 15;
  const int khalf = (lane >> 4) * 8;
  const int kbeg  = wave * KSLICE;

  const _Float16* ap = A  + (size_t)(mt * 16 + l15) * K_ + kbeg + khalf;
  const _Float16* bp = Bm + (size_t)(nt * 16 + l15) * K_ + kbeg + khalf;

  // ---- prologue: buffer 0 covers k [0, 128) ----
  v16h a00 = load_frag(ap + 0);
  v16h b00 = load_frag(bp + 0);
  v16h a01 = load_frag(ap + 32);
  v16h b01 = load_frag(bp + 32);
  v16h a02 = load_frag(ap + 64);
  v16h b02 = load_frag(bp + 64);
  v16h a03 = load_frag(ap + 96);
  v16h b03 = load_frag(bp + 96);
  v16h a10, b10, a11, b11, a12, b12, a13, b13;

  v8f acc0 = {}, acc1 = {}, acc2 = {}, acc3 = {};

  int k0 = 0;
#pragma unroll 1
  for (; k0 < KSLICE - 256; k0 += 256) {
    // issue buffer-1 loads [k0+128, k0+256)
    a10 = load_frag(ap + k0 + 128);
    b10 = load_frag(bp + k0 + 128);
    a11 = load_frag(ap + k0 + 160);
    b11 = load_frag(bp + k0 + 160);
    a12 = load_frag(ap + k0 + 192);
    b12 = load_frag(bp + k0 + 192);
    a13 = load_frag(ap + k0 + 224);
    b13 = load_frag(bp + k0 + 224);
    // consume buffer 0 while buffer-1 loads are in flight
    acc0 = WMMA_F16(a00, b00, acc0);
    acc1 = WMMA_F16(a01, b01, acc1);
    acc2 = WMMA_F16(a02, b02, acc2);
    acc3 = WMMA_F16(a03, b03, acc3);
    // issue buffer-0 loads [k0+256, k0+384)
    a00 = load_frag(ap + k0 + 256);
    b00 = load_frag(bp + k0 + 256);
    a01 = load_frag(ap + k0 + 288);
    b01 = load_frag(bp + k0 + 288);
    a02 = load_frag(ap + k0 + 320);
    b02 = load_frag(bp + k0 + 320);
    a03 = load_frag(ap + k0 + 352);
    b03 = load_frag(bp + k0 + 352);
    // consume buffer 1 while buffer-0 loads are in flight
    acc0 = WMMA_F16(a10, b10, acc0);
    acc1 = WMMA_F16(a11, b11, acc1);
    acc2 = WMMA_F16(a12, b12, acc2);
    acc3 = WMMA_F16(a13, b13, acc3);
  }
  // ---- epilogue: k0 == KSLICE-256; buffer 0 holds [k0, k0+128) ----
  a10 = load_frag(ap + k0 + 128);
  b10 = load_frag(bp + k0 + 128);
  a11 = load_frag(ap + k0 + 160);
  b11 = load_frag(bp + k0 + 160);
  a12 = load_frag(ap + k0 + 192);
  b12 = load_frag(bp + k0 + 192);
  a13 = load_frag(ap + k0 + 224);
  b13 = load_frag(bp + k0 + 224);
  acc0 = WMMA_F16(a00, b00, acc0);
  acc1 = WMMA_F16(a01, b01, acc1);
  acc2 = WMMA_F16(a02, b02, acc2);
  acc3 = WMMA_F16(a03, b03, acc3);
  acc0 = WMMA_F16(a10, b10, acc0);
  acc1 = WMMA_F16(a11, b11, acc1);
  acc2 = WMMA_F16(a12, b12, acc2);
  acc3 = WMMA_F16(a13, b13, acc3);

  v8f acc = (acc0 + acc1) + (acc2 + acc3);

  if (wave != 0) red[wave - 1][lane] = acc;
  __syncthreads();

  if (wave == 0) {
    acc = acc + red[0][lane];
    acc = acc + red[1][lane];
    acc = acc + red[2][lane];
    // C/D layout: lanes 0-15 -> M = r, lanes 16-31 -> M = 8 + r; N = lane&15
    const int r0 = (lane >> 4) * 8;
    float* out = C + (size_t)(mt * 16 + r0) * O_ + nt * 16 + l15;
#pragma unroll
    for (int r = 0; r < 8; ++r)
      out[(size_t)r * O_] = acc[r];
  }
}

// ---------------------------------------------------------------------------
extern "C" void kernel_launch(void* const* d_in, const int* in_sizes, int n_in,
                              void* d_out, int out_size, void* d_ws, size_t ws_size,
                              hipStream_t stream) {
  (void)in_sizes; (void)n_in; (void)out_size; (void)ws_size;
  const float* x   = (const float*)d_in[0];
  const float* pos = (const float*)d_in[1];
  const float* val = (const float*)d_in[2];
  float* out = (float*)d_out;

  _Float16* Aws = (_Float16*)d_ws;                                   // 2 MB
  _Float16* Bws = (_Float16*)((char*)d_ws +
                              (size_t)B_ * K_ * sizeof(_Float16));   // 8 MB

  build_basis<<<(B_ * I_) / 256, 256, 0, stream>>>(x, pos, Aws);
  pack_values<<<(I_ * O_) / 256, 256, 0, stream>>>(val, Bws);
  wmma_gemm<<<256, 128, 0, stream>>>(Aws, Bws, out);
}
